// DeepJ_52097953301251
// MI455X (gfx1250) — matrix operand order repaired and checked by reading, not compiled, not run on hardware
//
#include <hip/hip_runtime.h>
#include <hip/hip_bf16.h>
#include <stdint.h>

// ---------------- constants ----------------
#define B_      1024
#define NNOTES  48
#define NB      49152        // NNOTES * B_
#define TU      256          // T_UNITS
#define TIN     50
#define KT0     320          // pad32(50 + 256)
#define KT1     512          // 256 + 256
#define NU      128          // N_UNITS
#define KN0     416          // pad32(257 + 128)
#define KN1     256          // 128 + 128

typedef __bf16 v16bf __attribute__((ext_vector_type(16)));
typedef float  v8f   __attribute__((ext_vector_type(8)));

__device__ __forceinline__ unsigned short f2bf_bits(float x) {
    unsigned int u = __float_as_uint(x);
    unsigned int r = (u + 0x7FFFu + ((u >> 16) & 1u)) >> 16;
    return (unsigned short)r;
}
__device__ __forceinline__ float sigf(float x) {
    return 1.0f / (1.0f + __expf(-x));
}

union FragU { uint4 q[2]; v16bf v; };

// A fragment, this code's contiguous k order (same as pack_w): lane<16: row=m0+lane, elems 0..15 -> K+0..15
//             lane>=16: row=m0+lane-16, elems 0..15 -> K+16..31
__device__ __forceinline__ void loadA(FragU& f, const unsigned short* p) {
    f.q[0] = *(const uint4*)(p);
    f.q[1] = *(const uint4*)(p + 8);
}
// B fragment is pre-packed: 32 lanes x 16 bf16 contiguous per 32x16 K-tile
__device__ __forceinline__ void loadB(FragU& f, const unsigned short* p) {
    f.q[0] = *(const uint4*)(p);
    f.q[1] = *(const uint4*)(p + 8);
}

// ---------------- weight repack: [W_ih; W_hh] -> fragment-major bf16 ----------------
// B-fragment, contiguous k order shared with the A loads: lane<16 -> col=lane, K=kt*32+e ; lane>=16 -> col=lane-16, K=kt*32+16+e
__global__ void pack_w(const float* __restrict__ Wih, int Kih,
                       const float* __restrict__ Whh, int Khh,
                       int Kpad, int fourH, unsigned short* __restrict__ out) {
    int idx = blockIdx.x * blockDim.x + threadIdx.x;
    int total = Kpad * fourH;
    if (idx >= total) return;
    int n = idx % fourH;
    int k = idx / fourH;
    float v = 0.0f;
    if (k < Kih)            v = Wih[(size_t)n * Kih + k];
    else if (k < Kih + Khh) v = Whh[(size_t)n * Khh + (k - Kih)];
    int kT = k >> 5, kr = k & 31, cT = n >> 4, cr = n & 15;
    int lane = cr + ((kr & 16) ? 16 : 0);
    int e    = kr & 15;
    int nCT  = fourH >> 4;
    out[(((size_t)(kT * nCT + cT)) * 32 + lane) * 16 + e] = f2bf_bits(v);
}

__global__ void combine_bias(const float* __restrict__ bih, const float* __restrict__ bhh,
                             float* __restrict__ out, int n) {
    int i = blockIdx.x * blockDim.x + threadIdx.x;
    if (i < n) out[i] = bih[i] + bhh[i];
}

// ---------------- feature build: A_T0 = [pp|pc|vic|chord | h_prev(L0) | pad] as bf16 ----------------
__global__ void build_AT0(const float* __restrict__ note, const float* __restrict__ states_h,
                          unsigned short* __restrict__ A) {
    int row = blockIdx.x;            // 0..NB-1  (n*B + b)
    int k   = threadIdx.x;           // 0..KT0-1
    int n = row >> 10, b = row & 1023;
    float val = 0.0f;
    if (k == 0) {
        val = (float)n * (1.0f / 48.0f);
    } else if (k < 13) {
        val = ((n % 12) == (k - 1)) ? 1.0f : 0.0f;
    } else if (k < 38) {
        int idx = n + (k - 13);      // xpad index
        val = (idx >= 12 && idx < 60) ? note[b * NNOTES + (idx - 12)] : 0.0f;
    } else if (k < 50) {
        int j = k - 38;
        float s = 0.0f;
        #pragma unroll
        for (int o = 0; o < 4; ++o) s += note[b * NNOTES + j * 4 + o];
        val = s;
    } else if (k < 306) {
        int t = k - 50;
        val = states_h[((size_t)(n * 2) * B_ + b) * TU + t];   // layer 0 h
    }
    A[(size_t)row * KT0 + k] = f2bf_bits(val);
}

// A_T1 cols 256..511 = states_h layer1 (cols 0..255 written by time-L0 epilogue)
__global__ void build_AT1(const float* __restrict__ states_h, unsigned short* __restrict__ A) {
    int row = blockIdx.x;
    int k   = threadIdx.x;           // 0..255
    int n = row >> 10, b = row & 1023;
    float val = states_h[((size_t)(n * 2 + 1) * B_ + b) * TU + k];
    A[(size_t)row * KT1 + 256 + k] = f2bf_bits(val);
}

// A_N0 per-step slabs: col 256 = cond, cols 257..415 zero-init (257..384 later overwritten by note-L0)
__global__ void init_note(const float* __restrict__ targets, unsigned short* __restrict__ AN0) {
    int slab = blockIdx.x >> 10;     // t = 0..47
    int b    = blockIdx.x & 1023;
    int col  = 256 + threadIdx.x;    // 256..415
    float val = 0.0f;
    if (threadIdx.x == 0 && slab > 0) val = targets[b * NNOTES + slab - 1];
    AN0[((size_t)slab * B_ + b) * KN0 + col] = f2bf_bits(val);
}

// ---------------- fused WMMA LSTM cell ----------------
// One wave computes a 32(row) x 16(hidden-unit) tile: 2 A fragments x 4 gate B fragments
// -> 8 WMMA accumulators. Fragment loads are register double-buffered (ping-pong) so the
// next K-step's 12 global_load_b128 are in flight while the current 8 WMMAs execute.
// All pointers are scalarized (no pointer arrays) so InferAddressSpaces keeps them
// global (global_load_b128 / LOADcnt) instead of demoting to flat (DScnt).
// Addressing for fp32 state tensors: base[(row>>10)*sn + (row&1023)*sb + col]
__global__ __launch_bounds__(32)
void lstm_wmma_fused(const unsigned short* __restrict__ A, int lda,
                     const unsigned short* __restrict__ Wp, int kTiles, int hTiles, int H,
                     const float* __restrict__ bias,
                     const float* __restrict__ cprev, int cp_sn, int cp_sb,
                     float* __restrict__ cout, int co_sn, int co_sb,
                     float* __restrict__ hout, int ho_sn, int ho_sb,
                     unsigned short* __restrict__ hb1, int hb1_stride, int hb1_off,
                     unsigned short* __restrict__ hb2, int hb2_stride, int hb2_off) {
    const int lane    = threadIdx.x;
    const int halfSel = lane >> 4;
    const int l15     = lane & 15;
    const int tile = blockIdx.x;
    const int hT = tile % hTiles;
    const int mB = tile / hTiles;
    const int m0 = mB * 32, h0 = hT * 16;
    const int nColTiles = hTiles * 4;
    const size_t bStep = (size_t)nColTiles * 32 * 16;   // packed elements per K-tile

    v8f acc[2][4] = {};   // [row-subtile][gate i,f,g,o]

    const unsigned short* aPtr0 = A + (size_t)(m0 + l15) * lda + (halfSel ? 16 : 0);
    const unsigned short* aPtr1 = aPtr0 + (size_t)16 * lda;
    const unsigned short* bp0 = Wp + (((size_t)(0 * hTiles + hT)) * 32 + lane) * 16;
    const unsigned short* bp1 = Wp + (((size_t)(1 * hTiles + hT)) * 32 + lane) * 16;
    const unsigned short* bp2 = Wp + (((size_t)(2 * hTiles + hT)) * 32 + lane) * 16;
    const unsigned short* bp3 = Wp + (((size_t)(3 * hTiles + hT)) * 32 + lane) * 16;

    FragU a00, a01, a10, a11;
    FragU b00, b01, b02, b03, b10, b11, b12, b13;

    loadA(a00, aPtr0);
    loadA(a01, aPtr1);
    loadB(b00, bp0);
    loadB(b01, bp1);
    loadB(b02, bp2);
    loadB(b03, bp3);

    for (int kt = 0; kt < kTiles; kt += 2) {
        const bool hasOdd = (kt + 1 < kTiles);
        if (hasOdd) {                       // prefetch kt+1 into buffer 1
            aPtr0 += 32; aPtr1 += 32;
            bp0 += bStep; bp1 += bStep; bp2 += bStep; bp3 += bStep;
            loadA(a10, aPtr0);
            loadA(a11, aPtr1);
            loadB(b10, bp0);
            loadB(b11, bp1);
            loadB(b12, bp2);
            loadB(b13, bp3);
        }
        acc[0][0] = __builtin_amdgcn_wmma_f32_16x16x32_bf16(false, a00.v, false, b00.v, (short)0, acc[0][0], false, false);
        acc[1][0] = __builtin_amdgcn_wmma_f32_16x16x32_bf16(false, a01.v, false, b00.v, (short)0, acc[1][0], false, false);
        acc[0][1] = __builtin_amdgcn_wmma_f32_16x16x32_bf16(false, a00.v, false, b01.v, (short)0, acc[0][1], false, false);
        acc[1][1] = __builtin_amdgcn_wmma_f32_16x16x32_bf16(false, a01.v, false, b01.v, (short)0, acc[1][1], false, false);
        acc[0][2] = __builtin_amdgcn_wmma_f32_16x16x32_bf16(false, a00.v, false, b02.v, (short)0, acc[0][2], false, false);
        acc[1][2] = __builtin_amdgcn_wmma_f32_16x16x32_bf16(false, a01.v, false, b02.v, (short)0, acc[1][2], false, false);
        acc[0][3] = __builtin_amdgcn_wmma_f32_16x16x32_bf16(false, a00.v, false, b03.v, (short)0, acc[0][3], false, false);
        acc[1][3] = __builtin_amdgcn_wmma_f32_16x16x32_bf16(false, a01.v, false, b03.v, (short)0, acc[1][3], false, false);
        if (hasOdd) {
            if (kt + 2 < kTiles) {          // prefetch kt+2 into buffer 0
                aPtr0 += 32; aPtr1 += 32;
                bp0 += bStep; bp1 += bStep; bp2 += bStep; bp3 += bStep;
                loadA(a00, aPtr0);
                loadA(a01, aPtr1);
                loadB(b00, bp0);
                loadB(b01, bp1);
                loadB(b02, bp2);
                loadB(b03, bp3);
            }
            acc[0][0] = __builtin_amdgcn_wmma_f32_16x16x32_bf16(false, a10.v, false, b10.v, (short)0, acc[0][0], false, false);
            acc[1][0] = __builtin_amdgcn_wmma_f32_16x16x32_bf16(false, a11.v, false, b10.v, (short)0, acc[1][0], false, false);
            acc[0][1] = __builtin_amdgcn_wmma_f32_16x16x32_bf16(false, a10.v, false, b11.v, (short)0, acc[0][1], false, false);
            acc[1][1] = __builtin_amdgcn_wmma_f32_16x16x32_bf16(false, a11.v, false, b11.v, (short)0, acc[1][1], false, false);
            acc[0][2] = __builtin_amdgcn_wmma_f32_16x16x32_bf16(false, a10.v, false, b12.v, (short)0, acc[0][2], false, false);
            acc[1][2] = __builtin_amdgcn_wmma_f32_16x16x32_bf16(false, a11.v, false, b12.v, (short)0, acc[1][2], false, false);
            acc[0][3] = __builtin_amdgcn_wmma_f32_16x16x32_bf16(false, a10.v, false, b13.v, (short)0, acc[0][3], false, false);
            acc[1][3] = __builtin_amdgcn_wmma_f32_16x16x32_bf16(false, a11.v, false, b13.v, (short)0, acc[1][3], false, false);
        }
    }

    // C/D layout: lane<16 -> col=lane, rows +v ; lane>=16 -> col=lane-16, rows +8+v
    const int col = h0 + l15;
    const float bi = bias[col];
    const float bf_ = bias[H + col];
    const float bg = bias[2 * H + col];
    const float bo = bias[3 * H + col];

    #pragma unroll
    for (int mt = 0; mt < 2; ++mt) {
        const int rowBase = m0 + mt * 16 + (halfSel ? 8 : 0);
        #pragma unroll
        for (int v = 0; v < 8; ++v) {
            const int row = rowBase + v;
            const int hn_ = row >> 10, lb = row & 1023;
            const float cp = cprev[(size_t)hn_ * cp_sn + (size_t)lb * cp_sb + col];
            const float ig = sigf(acc[mt][0][v] + bi);
            const float fg = sigf(acc[mt][1][v] + bf_);
            const float gg = tanhf(acc[mt][2][v] + bg);
            const float og = sigf(acc[mt][3][v] + bo);
            const float cn = fg * cp + ig * gg;
            const float hv = og * tanhf(cn);
            cout[(size_t)hn_ * co_sn + (size_t)lb * co_sb + col] = cn;
            if (hout) hout[(size_t)hn_ * ho_sn + (size_t)lb * ho_sb + col] = hv;
            const unsigned short hb = f2bf_bits(hv);
            hb1[(size_t)row * hb1_stride + hb1_off + col] = hb;
            if (hb2) hb2[(size_t)row * hb2_stride + hb2_off + col] = hb;
        }
    }
}

// ---------------- output head: y[b] = sigmoid(h1 . outW + outb) ----------------
__global__ void out_gemv(const float* __restrict__ h1, const float* __restrict__ outW,
                         const float* __restrict__ outb, float* __restrict__ out, int t) {
    int b = blockIdx.x * blockDim.x + threadIdx.x;
    if (b >= B_) return;
    float s = outb[0];
    #pragma unroll 8
    for (int k = 0; k < NU; ++k) s += h1[b * NU + k] * outW[k];
    out[b * NNOTES + t] = sigf(s);
}

// ---------------- host ----------------
extern "C" void kernel_launch(void* const* d_in, const int* in_sizes, int n_in,
                              void* d_out, int out_size, void* d_ws, size_t ws_size,
                              hipStream_t stream) {
    const float* note    = (const float*)d_in[0];
    const float* targets = (const float*)d_in[1];
    const float* sh      = (const float*)d_in[2];
    const float* sc      = (const float*)d_in[3];
    const float* tWih0 = (const float*)d_in[4];
    const float* tWhh0 = (const float*)d_in[5];
    const float* tbih0 = (const float*)d_in[6];
    const float* tbhh0 = (const float*)d_in[7];
    const float* tWih1 = (const float*)d_in[8];
    const float* tWhh1 = (const float*)d_in[9];
    const float* tbih1 = (const float*)d_in[10];
    const float* tbhh1 = (const float*)d_in[11];
    const float* nWih0 = (const float*)d_in[12];
    const float* nWhh0 = (const float*)d_in[13];
    const float* nbih0 = (const float*)d_in[14];
    const float* nbhh0 = (const float*)d_in[15];
    const float* nWih1 = (const float*)d_in[16];
    const float* nWhh1 = (const float*)d_in[17];
    const float* nbih1 = (const float*)d_in[18];
    const float* nbhh1 = (const float*)d_in[19];
    const float* outW  = (const float*)d_in[20];
    const float* outb  = (const float*)d_in[21];

    float* outp = (float*)d_out;
    float* yOut = outp;                                   // (B, N)
    float* nhOut = outp + 49152;                          // (N,2,B,TU)
    float* ncOut = outp + 49152 + 25165824;               // (N,2,B,TU)

    uint8_t* base = (uint8_t*)d_ws;
    size_t off = 0;
    auto alloc = [&](size_t bytes) -> uint8_t* {
        uint8_t* r = base + off;
        off += (bytes + 255) & ~(size_t)255;
        return r;
    };
    unsigned short* WT0p = (unsigned short*)alloc((size_t)10 * 64 * 512 * 2);
    unsigned short* WT1p = (unsigned short*)alloc((size_t)16 * 64 * 512 * 2);
    unsigned short* WN0p = (unsigned short*)alloc((size_t)13 * 32 * 512 * 2);
    unsigned short* WN1p = (unsigned short*)alloc((size_t)8  * 32 * 512 * 2);
    float* bT0 = (float*)alloc(1024 * 4);
    float* bT1 = (float*)alloc(1024 * 4);
    float* bN0 = (float*)alloc(512 * 4);
    float* bN1 = (float*)alloc(512 * 4);
    unsigned short* AT0 = (unsigned short*)alloc((size_t)NB * KT0 * 2);
    unsigned short* AT1 = (unsigned short*)alloc((size_t)NB * KT1 * 2);
    unsigned short* AN0 = (unsigned short*)alloc((size_t)49 * B_ * KN0 * 2);  // 48 steps + 1 spill slab
    unsigned short* AN1 = (unsigned short*)alloc((size_t)2 * B_ * KN1 * 2);   // ping-pong
    float* cN0 = (float*)alloc((size_t)B_ * NU * 4);
    float* cN1 = (float*)alloc((size_t)B_ * NU * 4);
    float* h1n = (float*)alloc((size_t)B_ * NU * 4);
    (void)ws_size; (void)in_sizes; (void)n_in; (void)out_size;

    // zero recurrent-state buffers (graph-capturable)
    hipMemsetAsync(AN1, 0, (size_t)2 * B_ * KN1 * 2, stream);
    hipMemsetAsync(cN0, 0, (size_t)B_ * NU * 4, stream);
    hipMemsetAsync(cN1, 0, (size_t)B_ * NU * 4, stream);

    // weight / bias prep
    pack_w<<<(KT0 * 1024 + 255) / 256, 256, 0, stream>>>(tWih0, TIN, tWhh0, TU, KT0, 1024, WT0p);
    pack_w<<<(KT1 * 1024 + 255) / 256, 256, 0, stream>>>(tWih1, TU,  tWhh1, TU, KT1, 1024, WT1p);
    pack_w<<<(KN0 * 512 + 255) / 256, 256, 0, stream>>>(nWih0, 257, nWhh0, NU, KN0, 512, WN0p);
    pack_w<<<(KN1 * 512 + 255) / 256, 256, 0, stream>>>(nWih1, NU,  nWhh1, NU, KN1, 512, WN1p);
    combine_bias<<<4, 256, 0, stream>>>(tbih0, tbhh0, bT0, 1024);
    combine_bias<<<4, 256, 0, stream>>>(tbih1, tbhh1, bT1, 1024);
    combine_bias<<<2, 256, 0, stream>>>(nbih0, nbhh0, bN0, 512);
    combine_bias<<<2, 256, 0, stream>>>(nbih1, nbhh1, bN1, 512);

    // activation build
    build_AT0<<<NB, KT0, 0, stream>>>(note, sh, AT0);
    build_AT1<<<NB, 256, 0, stream>>>(sh, AT1);
    init_note<<<NNOTES * B_, 160, 0, stream>>>(targets, AN0);

    const int SN = 2 * B_ * TU;   // (N,2,B,TU) row->n stride
    // ---- time axis layer 0 : M=NB, K=320, H=256 ----
    lstm_wmma_fused<<<(NB / 32) * (TU / 16), 32, 0, stream>>>(
        AT0, KT0, WT0p, KT0 / 32, TU / 16, TU, bT0,
        sc, SN, TU,                              // c_prev  = states_c layer0
        ncOut, SN, TU,                           // c_new  -> next_c layer0
        nhOut, SN, TU,                           // h_new  -> next_h layer0
        AT1, KT1, 0,                             // h bf16 -> A_T1 cols 0..255
        (unsigned short*)nullptr, 0, 0);
    // ---- time axis layer 1 : M=NB, K=512, H=256 ----
    lstm_wmma_fused<<<(NB / 32) * (TU / 16), 32, 0, stream>>>(
        AT1, KT1, WT1p, KT1 / 32, TU / 16, TU, bT1,
        sc + B_ * TU, SN, TU,
        ncOut + B_ * TU, SN, TU,
        nhOut + B_ * TU, SN, TU,
        AN0, KN0, 0,                             // feats bf16 -> A_N0 slab n, cols 0..255
        (unsigned short*)nullptr, 0, 0);

    // ---- note axis scan: 48 sequential steps ----
    for (int t = 0; t < NNOTES; ++t) {
        // layer 0: A = A_N0 slab t (feats|cond|h_prev|pad), M=1024, K=416, H=128
        lstm_wmma_fused<<<(B_ / 32) * (NU / 16), 32, 0, stream>>>(
            AN0 + (size_t)t * B_ * KN0, KN0, WN0p, KN0 / 32, NU / 16, NU, bN0,
            cN0, 0, NU,
            cN0, 0, NU,
            (float*)nullptr, 0, 0,
            AN0 + (size_t)(t + 1) * B_ * KN0, KN0, 257,   // h0 -> next step's slab
            AN1 + (size_t)(t & 1) * B_ * KN1, KN1, 0);    // h0 -> this step's L1 A
        // layer 1: A = A_N1 slab t&1 ([h0(t) | h1(t-1)]), K=256, H=128
        lstm_wmma_fused<<<(B_ / 32) * (NU / 16), 32, 0, stream>>>(
            AN1 + (size_t)(t & 1) * B_ * KN1, KN1, WN1p, KN1 / 32, NU / 16, NU, bN1,
            cN1, 0, NU,
            cN1, 0, NU,
            h1n, 0, NU,                                   // fp32 h1 for the head
            AN1 + (size_t)((t + 1) & 1) * B_ * KN1, KN1, NU,  // h1 -> next step's L1 A
            (unsigned short*)nullptr, 0, 0);
        out_gemv<<<(B_ + 255) / 256, 256, 0, stream>>>(h1n, outW, outb, yOut, t);
    }
}
